// QLSTMPlus_65481071400691
// MI455X (gfx1250) — matrix operand order repaired and checked by reading, not compiled, not run on hardware
//
#include <hip/hip_runtime.h>

typedef __attribute__((ext_vector_type(16))) _Float16 v16h;
typedef __attribute__((ext_vector_type(2)))  __fp16   v2fp;
typedef __attribute__((ext_vector_type(8)))  float    v8f;
typedef int v4i_ __attribute__((vector_size(16)));

// ---------------- problem constants ----------------
constexpr int T_   = 512;
constexpr int B_   = 128;
constexpr int D_   = 512;
constexpr int H_   = 64;
constexpr int NG   = 192;            // gates 1..3 only (forget gate is dead in the reference)
constexpr int MT   = 32;             // rows per block
constexpr int MTOT = T_ * B_;        // 65536
constexpr int MLAST = MTOT - B_;

// d_out layout (floats): h1 full, h0_last, c0_last, h1_last, c1_last
constexpr int O_H0L = MTOT * H_;
constexpr int O_C0L = O_H0L + B_ * H_;
constexpr int O_H1L = O_C0L + B_ * H_;
constexpr int O_C1L = O_H1L + B_ * H_;

// ---------------- async-to-LDS availability ----------------
#if defined(__has_builtin)
#if __has_builtin(__builtin_amdgcn_global_load_async_to_lds_b128)
#define QL_ASYNC 1
#endif
#endif

#define TO_GLB(p) ((v4i_*)(p))
#define TO_LDS(p) ((__attribute__((address_space(3))) v4i_*)(unsigned long long)(p))

__device__ __forceinline__ void ql_wait_async() {
#if defined(QL_ASYNC)
#if __has_builtin(__builtin_amdgcn_s_wait_asynccnt)
  __builtin_amdgcn_s_wait_asynccnt(0);
#else
  asm volatile("s_wait_asynccnt 0x0" ::: "memory");
#endif
#endif
}

union CvtU { _Float16 hh[2]; v2fp p2; unsigned int u; };
struct U2 { unsigned int x, y; };

// ---------------- prep: pack f16-pair weights [K/2][192], bias+theta -------
__global__ __launch_bounds__(256) void qlstm_prep(
    const float* __restrict__ W0, const float* __restrict__ b0, const float* __restrict__ th0,
    const float* __restrict__ W1, const float* __restrict__ b1, const float* __restrict__ th1,
    unsigned int* __restrict__ Wp0, unsigned int* __restrict__ Wp1,
    float* __restrict__ bth0, float* __restrict__ bth1)
{
  int idx = blockIdx.x * 256 + threadIdx.x;        // grid covers 256*192
  if (idx < 256 * NG) {                            // layer0: K=512 -> 256 pairs
    int p = idx / NG, c = idx % NG;                // c in [0,192) maps to gate col 64+c
    CvtU cv;
    cv.hh[0] = (_Float16)W0[(size_t)(64 + c) * (D_ + H_) + 2 * p];
    cv.hh[1] = (_Float16)W0[(size_t)(64 + c) * (D_ + H_) + 2 * p + 1];
    Wp0[idx] = cv.u;
  }
  if (idx < 32 * NG) {                             // layer1: K=64 -> 32 pairs
    int p = idx / NG, c = idx % NG;
    CvtU cv;
    cv.hh[0] = (_Float16)W1[(size_t)(64 + c) * (H_ + H_) + 2 * p];
    cv.hh[1] = (_Float16)W1[(size_t)(64 + c) * (H_ + H_) + 2 * p + 1];
    Wp1[idx] = cv.u;
  }
  if (idx < 256) {
    bth0[idx] = b0[idx] + th0[idx];
    bth1[idx] = b1[idx] + th1[idx];
  }
}

// ---------------- fused GEMM (WMMA f16) + quantum-gate activation + LN -----
struct Phase1 {
  _Float16     sXa[2][MT][40];       // A K-slab, double buffered (f16 pairs)
  unsigned int sWp[2][16][NG];       // W K-slab: 16 K-pairs x 192 cols, double buffered
};
struct Phase2 {
  float sZ[MT][NG + 4];              // z tile 32 x 192
  float sSG[3][MT][64];              // processed gates i, g, o
  float sH[MT][66];                  // raw h before LN
  float sMu[MT];
  float sRs[MT];
};
union SMem { Phase1 p1; Phase2 p2; };

// L0=true: A = f32 inputs (K=512), write h0 as f16 + last h0/c0.
// L0=false: A = f16 h0 (K=64), write full f32 h1 + last h1/c1.
template <bool L0>
__global__ __launch_bounds__(256) void qlstm_gemm_act(
    const float* __restrict__ Af32,
    const _Float16* __restrict__ Af16,
    const unsigned int* __restrict__ Wp, // [K/2][192] packed f16 pairs
    const float* __restrict__ bth,       // [256] bias+theta (full 4-gate layout)
    const float* __restrict__ gamma, const float* __restrict__ beta,
    float* __restrict__ hfull_out,
    _Float16* __restrict__ h16_out,
    float* __restrict__ hlast_out,
    float* __restrict__ clast_out)
{
  __shared__ SMem sm;
  constexpr int K      = L0 ? D_ : H_;
  constexpr int KSTEPS = K / 32;
  const int tid   = threadIdx.x;
  const int wave  = tid >> 5;
  const int lane  = tid & 31;
  const int h     = lane >> 4;            // lane half
  const int ml    = lane & 15;
  const int mBase = blockIdx.x * MT;
  const int sub   = wave >> 2;            // M subtile (0: rows 0-15, 1: rows 16-31)
  const int c0    = (wave & 3) * 48;      // this wave's first of 3 contiguous col tiles

  // A staging geometry: thread handles 2 consecutive f16-pairs (1/8 of a 32-K row)
  const int arow = tid >> 3;              // 32 rows
  const int app  = (tid & 7) * 2;         // even pair index in [0,16)

  // W slab per k-step: 16 K-pairs x 192 dwords = 12 KB, contiguous in Wp
#if defined(QL_ASYNC)
  {
    const char* g = (const char*)Wp;      // slab 0
    char* l = (char*)&sm.p1.sWp[0][0][0];
#pragma unroll
    for (int r = 0; r < 3; ++r) {
      int chunk = (tid + r * 256) * 16;
      __builtin_amdgcn_global_load_async_to_lds_b128(TO_GLB(g + chunk), TO_LDS(l + chunk), 0, 0);
    }
  }
#endif

  v8f acc[3] = {};

  for (int kb = 0; kb < KSTEPS; ++kb) {
    const int buf = kb & 1;

    // ---- stage A K-slab (32 rows x 32 K) as packed f16 pairs ----
    {
      U2 w;
      if constexpr (L0) {
        const float4 v = *(const float4*)(Af32 + (size_t)(mBase + arow) * K + kb * 32 + 2 * app);
        CvtU a, b;
        a.p2 = __builtin_amdgcn_cvt_pkrtz(v.x, v.y);
        b.p2 = __builtin_amdgcn_cvt_pkrtz(v.z, v.w);
        w.x = a.u; w.y = b.u;
      } else {
        w = *(const U2*)(Af16 + (size_t)(mBase + arow) * K + kb * 32 + 2 * app);
      }
      *(U2*)&sm.p1.sXa[buf][arow][2 * app] = w;
    }

#if defined(QL_ASYNC)
    ql_wait_async();                     // W slab kb has landed (issued last iter)
    __syncthreads();                     // + A slab kb visible to all
    if (kb + 1 < KSTEPS) {               // prefetch next W slab into other buffer
      const char* g = (const char*)Wp + (size_t)(kb + 1) * 12288;
      char* l = (char*)&sm.p1.sWp[buf ^ 1][0][0];
#pragma unroll
      for (int r = 0; r < 3; ++r) {
        int chunk = (tid + r * 256) * 16;
        __builtin_amdgcn_global_load_async_to_lds_b128(TO_GLB(g + chunk), TO_LDS(l + chunk), 0, 0);
      }
    }
    const int wb = buf;
#else
    __syncthreads();
    {                                     // synchronous W slab copy
      unsigned int* dst = &sm.p1.sWp[0][0][0];
#pragma unroll
      for (int r = 0; r < 12; ++r) {
        int i = tid + r * 256;            // 3072 dwords
        dst[i] = Wp[(size_t)kb * 3072 + i];
      }
    }
    __syncthreads();
    const int wb = 0;
#endif

    // ---- build fragments ----
    union F { v16h v; unsigned int u[8]; } af, bf0, bf1, bf2;
#pragma unroll
    for (int j = 0; j < 8; ++j) {
      // A 16x32 f16 layout: h=0 -> K 0..7 (v0-3), 16..23 (v4-7); h=1 -> 8..15, 24..31
      int pa = (j < 4 ? j : j + 4) + 4 * h;
      af.u[j] = *(const unsigned int*)&sm.p1.sXa[buf][sub * 16 + ml][2 * pa];
      // B 32x16 layout (matches A): h=0 -> K 0..7, 16..23; h=1 -> 8..15, 24..31
      int pb = (j < 4 ? j : j + 4) + 4 * h;
      bf0.u[j] = sm.p1.sWp[wb][pb][c0 + ml];
      bf1.u[j] = sm.p1.sWp[wb][pb][c0 + 16 + ml];
      bf2.u[j] = sm.p1.sWp[wb][pb][c0 + 32 + ml];
    }
    acc[0] = __builtin_amdgcn_wmma_f32_16x16x32_f16(false, af.v, false, bf0.v, (short)0, acc[0], false, false);
    acc[1] = __builtin_amdgcn_wmma_f32_16x16x32_f16(false, af.v, false, bf1.v, (short)0, acc[1], false, false);
    acc[2] = __builtin_amdgcn_wmma_f32_16x16x32_f16(false, af.v, false, bf2.v, (short)0, acc[2], false, false);
  }

  __syncthreads();                        // all fragment reads done; Phase2 overlays Phase1

  // ---- spill C tiles: VGPR v, lanes 0-15 -> (M=v, N=lane); 16-31 -> (M=v+8) ----
#pragma unroll
  for (int v = 0; v < 8; ++v) {
#pragma unroll
    for (int j = 0; j < 3; ++j)
      sm.p2.sZ[sub * 16 + v + 8 * h][c0 + j * 16 + ml] = acc[j][v];
  }
  __syncthreads();

  // ---- quantum gate: q_n = prod_{i<=n} cos(z_i + b_i + th_i) ----
  if (tid < 96) {
    int row = tid & 31;
    int zg  = tid >> 5;                  // 0,1,2 -> gates input, update, output
    float prod = 1.0f;
    for (int n = 0; n < 64; ++n) {
      float z = sm.p2.sZ[row][zg * 64 + n] + bth[64 + zg * 64 + n];
      prod *= __cosf(z);
      float s = 1.0f / (1.0f + __expf(-prod));     // sigmoid(q)
      float val = s;
      if (zg == 1) {                                // tanh(sigmoid(q))
        float e = __expf(-2.0f * s);
        val = (1.0f - e) / (1.0f + e);
      }
      sm.p2.sSG[zg][row][n] = val;
    }
  }
  __syncthreads();

  // ---- c = i*g, h = o*tanh(c) ----
  {
    int idx = tid;
#pragma unroll
    for (int rep = 0; rep < 8; ++rep, idx += 256) {
      int row = idx >> 6, n = idx & 63;
      float ig = sm.p2.sSG[0][row][n];
      float gg = sm.p2.sSG[1][row][n];
      float og = sm.p2.sSG[2][row][n];
      float c  = ig * gg;
      float e  = __expf(-2.0f * c);
      float hh = og * (1.0f - e) / (1.0f + e);
      sm.p2.sH[row][n] = hh;
      int m = mBase + row;
      if (m >= MLAST) clast_out[(m - MLAST) * 64 + n] = c;
    }
  }
  __syncthreads();

  // ---- LayerNorm stats over H=64 ----
  if (tid < MT) {
    float s = 0.f;
    for (int n = 0; n < 64; ++n) s += sm.p2.sH[tid][n];
    float mu = s * (1.0f / 64.0f);
    float v = 0.f;
    for (int n = 0; n < 64; ++n) { float d = sm.p2.sH[tid][n] - mu; v += d * d; }
    sm.p2.sMu[tid] = mu;
    sm.p2.sRs[tid] = rsqrtf(v * (1.0f / 64.0f) + 1e-5f);
  }
  __syncthreads();

  // ---- apply LN, write outputs ----
  {
    int idx = tid;
#pragma unroll
    for (int rep = 0; rep < 8; ++rep, idx += 256) {
      int row = idx >> 6, n = idx & 63;
      float hln = (sm.p2.sH[row][n] - sm.p2.sMu[row]) * sm.p2.sRs[row] * gamma[n] + beta[n];
      size_t m = (size_t)(mBase + row);
      if constexpr (L0) {
        h16_out[m * 64 + n] = (_Float16)hln;
      } else {
        hfull_out[m * 64 + n] = hln;
      }
      if ((int)m >= MLAST) hlast_out[((int)m - MLAST) * 64 + n] = hln;
    }
  }
}

extern "C" void kernel_launch(void* const* d_in, const int* in_sizes, int n_in,
                              void* d_out, int out_size, void* d_ws, size_t ws_size,
                              hipStream_t stream) {
  const float* x   = (const float*)d_in[0];
  const float* W0  = (const float*)d_in[1];
  const float* b0  = (const float*)d_in[2];
  const float* th0 = (const float*)d_in[3];
  const float* g0  = (const float*)d_in[4];
  const float* be0 = (const float*)d_in[5];
  const float* W1  = (const float*)d_in[6];
  const float* b1  = (const float*)d_in[7];
  const float* th1 = (const float*)d_in[8];
  const float* g1  = (const float*)d_in[9];
  const float* be1 = (const float*)d_in[10];
  float* out = (float*)d_out;

  char* ws = (char*)d_ws;
  unsigned int* Wp0 = (unsigned int*)(ws);                 // 256*192*4 = 192 KB
  unsigned int* Wp1 = (unsigned int*)(ws + 196608);        //  32*192*4 =  24 KB
  float*        bth0 = (float*)(ws + 221184);              // 1 KB
  float*        bth1 = (float*)(ws + 222208);              // 1 KB
  _Float16*     h0h  = (_Float16*)(ws + 223232);           // 65536*64*2 = 8 MB

  qlstm_prep<<<(256 * NG) / 256, 256, 0, stream>>>(
      W0, b0, th0, W1, b1, th1, Wp0, Wp1, bth0, bth1);

  // layer 0: K=512, A = f32 inputs; emit f16 h0 + last-step h0/c0 states
  qlstm_gemm_act<true><<<MTOT / MT, 256, 0, stream>>>(
      x, nullptr, Wp0, bth0, g0, be0,
      nullptr, h0h, out + O_H0L, out + O_C0L);

  // layer 1: K=64, A = f16 h0; emit full f32 h1 + last-step h1/c1 states
  qlstm_gemm_act<false><<<MTOT / MT, 256, 0, stream>>>(
      nullptr, h0h, Wp1, bth1, g1, be1,
      out, nullptr, out + O_H1L, out + O_C1L);
}
